// E_GCL_vel_10599979287281
// MI455X (gfx1250) — matrix operand order repaired and checked by reading, not compiled, not run on hardware
//
#include <hip/hip_runtime.h>
#include <hip/hip_bf16.h>

// ---------------------------------------------------------------------------
// EGNN (E_GCL_vel) for MI455X / gfx1250, f16 WMMA path.
// N=50000 nodes, E=1200000 edges, F=H=64, EDGE_IN=141 (padded to 160).
// All GEMMs run as v_wmma_f32_16x16x32_f16 with 16-row tiles per wave32.
// ---------------------------------------------------------------------------

typedef __attribute__((ext_vector_type(16))) _Float16 v16h;
typedef __attribute__((ext_vector_type(8)))  _Float16 v8h;
typedef __attribute__((ext_vector_type(8)))  float    v8f;

#define FDIM 64
#define HDIM 64
#define KIN_PAD 160   // 141 padded to 5 chunks of 32

// Packed-weight fragment offsets (in _Float16 units). One fragment =
// 32 lanes x 16 halves = 512 halves. Fragment (kchunk c, ntile t) at c*Nt+t.
#define WE1F 0        // 5 chunks x 4 tiles = 20 frags
#define WE2F 10240    // 2 x 4 = 8
#define WC1F 14336    // 2 x 4 = 8
#define WC2F 18432    // 2 x 1 = 2  (N padded 3->16)
#define WN1F 19456    // 4 x 4 = 16
#define WN2F 27648    // 2 x 4 = 8
#define WV1F 31744    // 2 x 4 = 8
#define WV2F 35840    // 2 x 1 = 2  (N padded 1->16)
#define WBYTES 73728  // 36864 halves

// ---------------------------------------------------------------------------
// helpers
// ---------------------------------------------------------------------------
__device__ __forceinline__ v16h cat8(v8h lo, v8h hi) {
  v16h r;
#pragma unroll
  for (int i = 0; i < 8; ++i) { r[i] = lo[i]; r[i + 8] = hi[i]; }
  return r;
}

// A-fragment (16x32, MxK, f16) from an LDS tile with row stride `stride`
// halves. Per ISA 7.12.2: lanes 0-15 rows M=lane, VGPR0..3 hold K=0..7
// (K+=8 for lanes 16-31), VGPR4..7 hold K=16..23 (K+=8 for lanes 16-31).
__device__ __forceinline__ v16h load_a_lds(const _Float16* base, int lane,
                                           int kbase, int stride) {
  const int m = lane & 15, hh = lane >> 4;
  const _Float16* p = base + m * stride + kbase + 8 * hh;
  v8h lo = *(const v8h*)p;          // K = kbase+8h .. +7
  v8h hi = *(const v8h*)(p + 16);   // K = kbase+16+8h .. +7
  return cat8(lo, hi);
}

// B-fragment (32x16, KxN, f16) from packed global weights: lane l holds
// column N=l&15, K = 8*(l>>4) + e (e<8), 16 + 8*(l>>4) + (e-8) (e>=8),
// stored lane-contiguous (2x b128 loads).
__device__ __forceinline__ v16h load_b_glob(const _Float16* wf, int frag,
                                            int lane) {
  const _Float16* p = wf + frag * 512 + lane * 16;
  v8h lo = *(const v8h*)p;
  v8h hi = *(const v8h*)(p + 8);
  return cat8(lo, hi);
}

__device__ __forceinline__ v8f wmma32(v16h a, v16h b, v8f c) {
  return __builtin_amdgcn_wmma_f32_16x16x32_f16(false, a, false, b,
                                                (short)0, c, false, false);
}

// C/D fragment element r of lane l maps to M = r + 8*(l>>4), N = l&15.
__device__ __forceinline__ void store_relu_tile(_Float16* tile, v8f acc,
                                                float bias, int lane, int nt) {
  const int m16 = lane & 15, hh = lane >> 4;
#pragma unroll
  for (int r = 0; r < 8; ++r) {
    float v = acc[r] + bias;
    v = v > 0.f ? v : 0.f;
    tile[(r + 8 * hh) * 64 + nt * 16 + m16] = (_Float16)v;
  }
}

__device__ __forceinline__ void lds_fence() {
  asm volatile("s_wait_dscnt 0x0" ::: "memory");
  __builtin_amdgcn_wave_barrier();
}

// ---------------------------------------------------------------------------
// weight packing: W is (K x Ncols) row-major fp32; emit f16 fragments.
// ---------------------------------------------------------------------------
__global__ void pack_weight(const float* __restrict__ W, _Float16* __restrict__ dst,
                            int K, int Ncols, int Kchunks, int Ntiles) {
  int id = blockIdx.x * blockDim.x + threadIdx.x;
  int total = Kchunks * Ntiles * 512;
  if (id >= total) return;
  int e = id & 15;
  int lane = (id >> 4) & 31;
  int frag = id >> 9;
  int c = frag / Ntiles, t = frag % Ntiles;
  int k = c * 32 + 8 * (lane >> 4) + (e < 8 ? e : e + 8);
  int n = t * 16 + (lane & 15);
  float v = (k < K && n < Ncols) ? W[k * Ncols + n] : 0.f;
  dst[frag * 512 + lane * 16 + e] = (_Float16)v;
}

__global__ void zero_f32(float* __restrict__ p, int n) {
  int id = blockIdx.x * blockDim.x + threadIdx.x;
  if (id < n) p[id] = 0.f;
}

// ---------------------------------------------------------------------------
// edge kernel: one wave = 16 edges. 38 WMMAs per wave tile.
// ---------------------------------------------------------------------------
__global__ __launch_bounds__(256) void egcl_edge_kernel(
    const float* __restrict__ h, const float* __restrict__ coord,
    const float* __restrict__ vel, const int* __restrict__ eidx, int E,
    const _Float16* __restrict__ wf, const float* __restrict__ be1,
    const float* __restrict__ be2, const float* __restrict__ bc1,
    float* __restrict__ agg, float* __restrict__ seg, float* __restrict__ cnt,
    int numTiles) {
  __shared__ __align__(16) _Float16 s_ein[8][16 * KIN_PAD];  // also ef2 later
  __shared__ __align__(16) _Float16 s_tile[8][16 * 64];
  __shared__ float s_diff[8][16 * 3];

  const int lane = threadIdx.x & 31;
  const int wid  = threadIdx.x >> 5;
  const int tile = blockIdx.x * 8 + wid;
  if (tile >= numTiles) return;

  const int base = tile * 16;
  const int m16 = lane & 15, hh = lane >> 4;
  const int* rowp = eidx;
  const int* colp = eidx + E;
  _Float16* ein   = s_ein[wid];
  _Float16* tileA = s_tile[wid];
  float*    diffb = s_diff[wid];

  // ---- stage e_in = [h[row] | h[col] | radial(7) | vel_rot(6) | pad] ----
  for (int m = 0; m < 16; ++m) {
    int e = base + m;
    int r = rowp[e], c = colp[e];
    float2 hr = ((const float2*)h)[r * 32 + lane];
    float2 hc = ((const float2*)h)[c * 32 + lane];
    union { _Float16 f[2]; unsigned u; } pr, pc;
    pr.f[0] = (_Float16)hr.x; pr.f[1] = (_Float16)hr.y;
    pc.f[0] = (_Float16)hc.x; pc.f[1] = (_Float16)hc.y;
    *(unsigned*)&ein[m * KIN_PAD + 2 * lane]      = pr.u;  // feats 0..63
    *(unsigned*)&ein[m * KIN_PAD + 64 + 2 * lane] = pc.u;  // feats 64..127
  }
  if (lane < 16) {
    int e = base + lane;
    int r = rowp[e], c = colp[e];
    float crx = coord[r * 3], cry = coord[r * 3 + 1], crz = coord[r * 3 + 2];
    float ccx = coord[c * 3], ccy = coord[c * 3 + 1], ccz = coord[c * 3 + 2];
    float dx = crx - ccx, dy = cry - ccy, dz = crz - ccz;
    float d2 = dx * dx + dy * dy + dz * dz;
    diffb[lane * 3]     = dx;
    diffb[lane * 3 + 1] = dy;
    diffb[lane * 3 + 2] = dz;
    float ext[13] = {crx, cry, crz, ccx, ccy, ccz, d2,
                     vel[r * 3], vel[r * 3 + 1], vel[r * 3 + 2],
                     vel[c * 3], vel[c * 3 + 1], vel[c * 3 + 2]};
    _Float16 vals[32];
#pragma unroll
    for (int i = 0; i < 32; ++i) vals[i] = (_Float16)0.f;
#pragma unroll
    for (int i = 0; i < 13; ++i) vals[i] = (_Float16)ext[i];
    unsigned* dst = (unsigned*)&ein[lane * KIN_PAD + 128];
#pragma unroll
    for (int i = 0; i < 16; ++i) {
      union { _Float16 f[2]; unsigned u; } p;
      p.f[0] = vals[2 * i]; p.f[1] = vals[2 * i + 1];
      dst[i] = p.u;
    }
  }
  lds_fence();

  // ---- edge MLP layer 1: 160x64, 5 K-chunks x 4 N-tiles ----
  v8f acc[4] = {v8f{}, v8f{}, v8f{}, v8f{}};
#pragma unroll
  for (int c = 0; c < 5; ++c) {
    v16h a = load_a_lds(ein, lane, c * 32, KIN_PAD);
#pragma unroll
    for (int nt = 0; nt < 4; ++nt)
      acc[nt] = wmma32(a, load_b_glob(wf + WE1F, c * 4 + nt, lane), acc[nt]);
  }
#pragma unroll
  for (int nt = 0; nt < 4; ++nt)
    store_relu_tile(tileA, acc[nt], be1[nt * 16 + m16], lane, nt);
  lds_fence();

  // ---- edge MLP layer 2: 64x64 -> ef (relu). agg atomics from f32 regs ----
#pragma unroll
  for (int nt = 0; nt < 4; ++nt) acc[nt] = v8f{};
#pragma unroll
  for (int c = 0; c < 2; ++c) {
    v16h a = load_a_lds(tileA, lane, c * 32, 64);
#pragma unroll
    for (int nt = 0; nt < 4; ++nt)
      acc[nt] = wmma32(a, load_b_glob(wf + WE2F, c * 4 + nt, lane), acc[nt]);
  }
  _Float16* ef = ein;  // staging dead; reuse as 16x64 ef tile
#pragma unroll
  for (int nt = 0; nt < 4; ++nt) {
    float bias = be2[nt * 16 + m16];
#pragma unroll
    for (int r = 0; r < 8; ++r) {
      int M = r + 8 * hh;
      float v = acc[nt][r] + bias;
      v = v > 0.f ? v : 0.f;
      ef[M * 64 + nt * 16 + m16] = (_Float16)v;
      atomicAdd(&agg[(long)rowp[base + M] * 64 + nt * 16 + m16], v);
    }
  }
  lds_fence();

  // ---- coord MLP layer 1: 64x64 relu -> p ----
#pragma unroll
  for (int nt = 0; nt < 4; ++nt) acc[nt] = v8f{};
#pragma unroll
  for (int c = 0; c < 2; ++c) {
    v16h a = load_a_lds(ef, lane, c * 32, 64);
#pragma unroll
    for (int nt = 0; nt < 4; ++nt)
      acc[nt] = wmma32(a, load_b_glob(wf + WC1F, c * 4 + nt, lane), acc[nt]);
  }
#pragma unroll
  for (int nt = 0; nt < 4; ++nt)
    store_relu_tile(tileA, acc[nt], bc1[nt * 16 + m16], lane, nt);
  lds_fence();

  // ---- coord MLP layer 2: 64x3 (N padded to 16) ----
  v8f acct = v8f{};
#pragma unroll
  for (int c = 0; c < 2; ++c) {
    v16h a = load_a_lds(tileA, lane, c * 32, 64);
    acct = wmma32(a, load_b_glob(wf + WC2F, c, lane), acct);
  }
  // trans = clip(coord_diff * t, +-100); segment-sum scatter
  if (m16 < 3) {
#pragma unroll
    for (int r = 0; r < 8; ++r) {
      int M = r + 8 * hh;
      float tr = diffb[M * 3 + m16] * acct[r];
      tr = fminf(fmaxf(tr, -100.f), 100.f);
      atomicAdd(&seg[(long)rowp[base + M] * 3 + m16], tr);
    }
  }
  if (lane < 16) atomicAdd(&cnt[rowp[base + lane]], 1.0f);
}

// ---------------------------------------------------------------------------
// node kernel: one wave = 16 nodes. Gate MLP + node MLP + coord assembly.
// ---------------------------------------------------------------------------
__global__ __launch_bounds__(256) void egcl_node_kernel(
    const float* __restrict__ h, const float* __restrict__ coord,
    const float* __restrict__ vel, const _Float16* __restrict__ wf,
    const float* __restrict__ bn1, const float* __restrict__ bn2,
    const float* __restrict__ bv1, const float* __restrict__ bv2,
    const float* __restrict__ agg, const float* __restrict__ seg,
    const float* __restrict__ cnt, float* __restrict__ out, int N,
    int numTiles) {
  __shared__ __align__(16) _Float16 s_xin[8][16 * 128];
  __shared__ __align__(16) _Float16 s_tile[8][16 * 64];
  __shared__ float s_gate[8][16];

  const int lane = threadIdx.x & 31;
  const int wid  = threadIdx.x >> 5;
  const int tile = blockIdx.x * 8 + wid;
  if (tile >= numTiles) return;

  const int base = tile * 16;
  const int m16 = lane & 15, hh = lane >> 4;
  _Float16* xin   = s_xin[wid];
  _Float16* tileA = s_tile[wid];
  float*    gateb = s_gate[wid];

  // ---- stage x = [h | agg] (16 x 128 f16) ----
  for (int m = 0; m < 16; ++m) {
    int node = base + m;
    float2 hv = ((const float2*)h)[node * 32 + lane];
    float2 av = ((const float2*)agg)[node * 32 + lane];
    union { _Float16 f[2]; unsigned u; } ph, pa;
    ph.f[0] = (_Float16)hv.x; ph.f[1] = (_Float16)hv.y;
    pa.f[0] = (_Float16)av.x; pa.f[1] = (_Float16)av.y;
    *(unsigned*)&xin[m * 128 + 2 * lane]      = ph.u;  // 0..63  h
    *(unsigned*)&xin[m * 128 + 64 + 2 * lane] = pa.u;  // 64..127 agg
  }
  lds_fence();

  // ---- velocity gate: relu(h@Wv1+bv1)@Wv2+bv2 (uses h chunks only) ----
  v8f acc[4] = {v8f{}, v8f{}, v8f{}, v8f{}};
#pragma unroll
  for (int c = 0; c < 2; ++c) {
    v16h a = load_a_lds(xin, lane, c * 32, 128);
#pragma unroll
    for (int nt = 0; nt < 4; ++nt)
      acc[nt] = wmma32(a, load_b_glob(wf + WV1F, c * 4 + nt, lane), acc[nt]);
  }
#pragma unroll
  for (int nt = 0; nt < 4; ++nt)
    store_relu_tile(tileA, acc[nt], bv1[nt * 16 + m16], lane, nt);
  lds_fence();

  v8f accg = v8f{};
#pragma unroll
  for (int c = 0; c < 2; ++c) {
    v16h a = load_a_lds(tileA, lane, c * 32, 64);
    accg = wmma32(a, load_b_glob(wf + WV2F, c, lane), accg);
  }
  if (m16 == 0) {
#pragma unroll
    for (int r = 0; r < 8; ++r) gateb[r + 8 * hh] = accg[r] + bv2[0];
  }
  lds_fence();

  // ---- coord_new = coord + seg/max(cnt,1) + gate*vel ----
  if (m16 < 3) {
#pragma unroll
    for (int r = 0; r < 8; ++r) {
      int M = r + 8 * hh;
      int node = base + M;
      float cn = coord[node * 3 + m16] +
                 seg[(long)node * 3 + m16] / fmaxf(cnt[node], 1.0f) +
                 gateb[M] * vel[node * 3 + m16];
      out[(long)N * 64 + (long)node * 3 + m16] = cn;
    }
  }

  // ---- node MLP layer 1: 128x64 relu ----
#pragma unroll
  for (int nt = 0; nt < 4; ++nt) acc[nt] = v8f{};
#pragma unroll
  for (int c = 0; c < 4; ++c) {
    v16h a = load_a_lds(xin, lane, c * 32, 128);
#pragma unroll
    for (int nt = 0; nt < 4; ++nt)
      acc[nt] = wmma32(a, load_b_glob(wf + WN1F, c * 4 + nt, lane), acc[nt]);
  }
  lds_fence();  // drain gate reads before overwriting tileA
#pragma unroll
  for (int nt = 0; nt < 4; ++nt)
    store_relu_tile(tileA, acc[nt], bn1[nt * 16 + m16], lane, nt);
  lds_fence();

  // ---- node MLP layer 2 + residual ----
#pragma unroll
  for (int nt = 0; nt < 4; ++nt) acc[nt] = v8f{};
#pragma unroll
  for (int c = 0; c < 2; ++c) {
    v16h a = load_a_lds(tileA, lane, c * 32, 64);
#pragma unroll
    for (int nt = 0; nt < 4; ++nt)
      acc[nt] = wmma32(a, load_b_glob(wf + WN2F, c * 4 + nt, lane), acc[nt]);
  }
#pragma unroll
  for (int nt = 0; nt < 4; ++nt) {
    float bias = bn2[nt * 16 + m16];
#pragma unroll
    for (int r = 0; r < 8; ++r) {
      int M = r + 8 * hh;
      int node = base + M;
      int f = nt * 16 + m16;
      out[(long)node * 64 + f] = h[(long)node * 64 + f] + acc[nt][r] + bias;
    }
  }
}

// ---------------------------------------------------------------------------
// launch
// ---------------------------------------------------------------------------
extern "C" void kernel_launch(void* const* d_in, const int* in_sizes, int n_in,
                              void* d_out, int out_size, void* d_ws,
                              size_t ws_size, hipStream_t stream) {
  const float* h     = (const float*)d_in[0];
  const float* coord = (const float*)d_in[1];
  const float* vel   = (const float*)d_in[2];
  const int*   eidx  = (const int*)d_in[3];
  const float* We1 = (const float*)d_in[4];
  const float* be1 = (const float*)d_in[5];
  const float* We2 = (const float*)d_in[6];
  const float* be2 = (const float*)d_in[7];
  const float* Wn1 = (const float*)d_in[8];
  const float* bn1 = (const float*)d_in[9];
  const float* Wn2 = (const float*)d_in[10];
  const float* bn2 = (const float*)d_in[11];
  const float* Wc1 = (const float*)d_in[12];
  const float* bc1 = (const float*)d_in[13];
  const float* Wc2 = (const float*)d_in[14];
  const float* Wv1 = (const float*)d_in[15];
  const float* bv1 = (const float*)d_in[16];
  const float* Wv2 = (const float*)d_in[17];
  const float* bv2 = (const float*)d_in[18];

  const int N = in_sizes[0] / 64;
  const int E = in_sizes[3] / 2;

  char* ws = (char*)d_ws;
  _Float16* wf = (_Float16*)ws;
  float* agg = (float*)(ws + WBYTES);
  float* seg = agg + (size_t)N * 64;
  float* cnt = seg + (size_t)N * 3;

  // pack weights into WMMA B-fragment order (f16)
  auto pk = [&](const float* W, int off, int K, int Nc, int Kc, int Nt) {
    int total = Kc * Nt * 512;
    pack_weight<<<(total + 255) / 256, 256, 0, stream>>>(W, wf + off, K, Nc,
                                                         Kc, Nt);
  };
  pk(We1, WE1F, 141, 64, 5, 4);
  pk(We2, WE2F, 64, 64, 2, 4);
  pk(Wc1, WC1F, 64, 64, 2, 4);
  pk(Wc2, WC2F, 64, 3, 2, 1);
  pk(Wn1, WN1F, 128, 64, 4, 4);
  pk(Wn2, WN2F, 64, 64, 2, 4);
  pk(Wv1, WV1F, 64, 64, 2, 4);
  pk(Wv2, WV2F, 64, 1, 2, 1);

  // zero accumulators (agg, seg, cnt are contiguous)
  int nzero = N * (64 + 3 + 1);
  zero_f32<<<(nzero + 255) / 256, 256, 0, stream>>>(agg, nzero);

  // edge pass: 16 edges per wave, 8 waves per block
  int eTiles = E / 16;
  egcl_edge_kernel<<<(eTiles + 7) / 8, 256, 0, stream>>>(
      h, coord, vel, eidx, E, wf, be1, be2, bc1, agg, seg, cnt, eTiles);

  // node pass
  int nTiles = N / 16;
  egcl_node_kernel<<<(nTiles + 7) / 8, 256, 0, stream>>>(
      h, coord, vel, wf, bn1, bn2, bv1, bv2, agg, seg, cnt, (float*)d_out, N,
      nTiles);
}
